// DeeperGCN_5463198401181
// MI455X (gfx1250) — matrix operand order repaired and checked by reading, not compiled, not run on hardware
//
#include <hip/hip_runtime.h>

#define NN 20000     // nodes
#define NE 640000    // edges
#define H  128
#define H2 256
#define NL 12
#define DOUT 16

typedef __attribute__((ext_vector_type(16))) _Float16 v16h;
typedef __attribute__((ext_vector_type(8)))  _Float16 v8h;
typedef __attribute__((ext_vector_type(8)))  float    v8f;

// ---- monotonic float<->uint encoding for atomicMax on floats ----
__device__ __forceinline__ unsigned fenc(float x) {
  unsigned b = __float_as_uint(x);
  return (b & 0x80000000u) ? ~b : (b | 0x80000000u);
}
__device__ __forceinline__ float fdec(unsigned e) {
  unsigned b = (e & 0x80000000u) ? (e & 0x7FFFFFFFu) : ~e;
  return __uint_as_float(b);
}

// ---------------- utility kernels ----------------
__global__ void k_zero(unsigned* __restrict__ p, int n) {
  int i = blockIdx.x * 256 + threadIdx.x;
  if (i < n) p[i] = 0u;
}

// h = x @ enc_node_w + enc_node_b   (N x 128, dot of 3)
__global__ void k_encode_nodes(const float* __restrict__ x, const float* __restrict__ w,
                               const float* __restrict__ b, float* __restrict__ h) {
  int tid = blockIdx.x * 256 + threadIdx.x;
  if (tid >= NN * H) return;
  int c = tid & (H - 1);
  int n = tid >> 7;
  h[tid] = b[c] + x[n * 3 + 0] * w[c] + x[n * 3 + 1] * w[H + c] + x[n * 3 + 2] * w[2 * H + c];
}

// hn = relu(LayerNorm(h)) ; one wave per node, 4 channels per lane
__global__ void k_prenorm(const float* __restrict__ h, const float* __restrict__ g,
                          const float* __restrict__ bb, float* __restrict__ hn) {
  int node = blockIdx.x * 8 + (threadIdx.x >> 5);
  int lane = threadIdx.x & 31;
  float4 v = *(const float4*)(h + (size_t)node * H + lane * 4);
  float s = v.x + v.y + v.z + v.w;
  float q = v.x * v.x + v.y * v.y + v.z * v.z + v.w * v.w;
  for (int off = 1; off < 32; off <<= 1) {
    s += __shfl_xor(s, off);
    q += __shfl_xor(q, off);
  }
  float mu = s * (1.0f / H);
  float var = q * (1.0f / H) - mu * mu;
  float rs = rsqrtf(var + 1e-5f);
  float4 gg = *(const float4*)(g + lane * 4);
  float4 bv = *(const float4*)(bb + lane * 4);
  float4 o;
  o.x = fmaxf((v.x - mu) * rs * gg.x + bv.x, 0.0f);
  o.y = fmaxf((v.y - mu) * rs * gg.y + bv.y, 0.0f);
  o.z = fmaxf((v.z - mu) * rs * gg.z + bv.z, 0.0f);
  o.w = fmaxf((v.w - mu) * rs * gg.w + bv.w, 0.0f);
  *(float4*)(hn + (size_t)node * H + lane * 4) = o;
}

// recompute encoded edge feature + message on the fly (keeps working set in L2)
__device__ __forceinline__ float edge_msg(int e, int c, int src,
                                          const float* __restrict__ hn,
                                          const float* __restrict__ eattr,
                                          const float* __restrict__ we,
                                          const float* __restrict__ be) {
  float ea = be[c] + eattr[e * 4 + 0] * we[c] + eattr[e * 4 + 1] * we[H + c] +
             eattr[e * 4 + 2] * we[2 * H + c] + eattr[e * 4 + 3] * we[3 * H + c];
  float m = hn[src * H + c] + ea;
  return fmaxf(m, 0.0f) + 1e-7f;
}

// pass 1: per-(dst,channel) segment max of logits via encoded atomicMax
__global__ void k_edge_max(const int* __restrict__ ei, const float* __restrict__ eattr,
                           const float* __restrict__ hn, const float* __restrict__ we,
                           const float* __restrict__ be, const float* __restrict__ tptr,
                           unsigned* __restrict__ mx) {
  int tid = blockIdx.x * 256 + threadIdx.x;
  if (tid >= NE * H) return;
  int c = tid & (H - 1);
  int e = tid >> 7;
  int src = ei[e];
  int dst = ei[NE + e];
  float m = edge_msg(e, c, src, hn, eattr, we, be);
  atomicMax(mx + dst * H + c, fenc(m * tptr[0]));
}

// pass 2: den += exp(logit-mx), num += m*exp(logit-mx)
__global__ void k_edge_sum(const int* __restrict__ ei, const float* __restrict__ eattr,
                           const float* __restrict__ hn, const float* __restrict__ we,
                           const float* __restrict__ be, const float* __restrict__ tptr,
                           const unsigned* __restrict__ mx, float* __restrict__ den,
                           float* __restrict__ num) {
  int tid = blockIdx.x * 256 + threadIdx.x;
  if (tid >= NE * H) return;
  int c = tid & (H - 1);
  int e = tid >> 7;
  int src = ei[e];
  int dst = ei[NE + e];
  float m = edge_msg(e, c, src, hn, eattr, we, be);
  float mxf = fdec(mx[dst * H + c]);
  float ex = __expf(m * tptr[0] - mxf);
  atomicAdd(den + dst * H + c, ex);
  atomicAdd(num + dst * H + c, m * ex);
}

// out_f16 = num/den + hn ; reset accumulators for the next layer
__global__ void k_node_finish(const float* __restrict__ hn, float* __restrict__ num,
                              float* __restrict__ den, unsigned* __restrict__ mx,
                              _Float16* __restrict__ outf16) {
  int tid = blockIdx.x * 256 + threadIdx.x;
  if (tid >= NN * H) return;
  float agg = num[tid] / (den[tid] + 1e-16f);
  outf16[tid] = (_Float16)(agg + hn[tid]);
  num[tid] = 0.0f;
  den[tid] = 0.0f;
  mx[tid] = 0u;
}

// pack fp32 weights into f16 WMMA B-operand swizzle:
//   lane l holds column n = l&15, K = (j>>3)*16 + (l>>4)*8 + (j&7) within each 32-K step.
__global__ void k_pack_weights(const float* __restrict__ w1, const float* __restrict__ w2,
                               _Float16* __restrict__ w1h, _Float16* __restrict__ w2h) {
  int tid = blockIdx.x * 256 + threadIdx.x;
  if (tid < NL * 32768) {               // W1: K=128 (4 ksteps), Nout=256 (16 ntiles)
    int q = tid & 32767;
    int layer = tid >> 15;
    int j = q & 15, lane = (q >> 4) & 31, ks = (q >> 9) & 3, nt = q >> 11;
    int k = ks * 32 + (j >> 3) * 16 + (lane >> 4) * 8 + (j & 7);
    int n = nt * 16 + (lane & 15);
    w1h[tid] = (_Float16)w1[layer * H * H2 + k * H2 + n];
  } else {
    int t2 = tid - NL * 32768;          // W2: K=256 (8 ksteps), Nout=128 (8 ntiles)
    if (t2 >= NL * 32768) return;
    int q = t2 & 32767;
    int layer = t2 >> 15;
    int j = q & 15, lane = (q >> 4) & 31, ks = (q >> 9) & 7, nt = q >> 12;
    int k = ks * 32 + (j >> 3) * 16 + (lane >> 4) * 8 + (j & 7);
    int n = nt * 16 + (lane & 15);
    w2h[t2] = (_Float16)w2[layer * H2 * H + k * H + n];
  }
}

// GEMM1: u = relu(LN(out_f16 @ W1 + b1)) -> f16.
// One wave owns 16 rows x 256 cols, computed as TWO halves of 8 column-tiles
// so only 8 v8f accumulators are live at once (no scratch spills). The A
// operand for all 4 K-steps is preloaded once and reused across both halves.
__global__ void k_gemm1_ln_relu(const _Float16* __restrict__ A, const _Float16* __restrict__ Bw,
                                const float* __restrict__ b1, const float* __restrict__ g1,
                                const float* __restrict__ bb1, _Float16* __restrict__ U) {
  __shared__ __align__(16) float T[2][16 * 264];   // +8 pad per row: bank-conflict free
  int wave = threadIdx.x >> 5, lane = threadIdx.x & 31;
  int mtile = blockIdx.x * 2 + wave;

  // preload A operand (16 rows x 128 K) for all 4 K-steps: 4 x v16h = 32 VGPRs
  const v8h* arow = (const v8h*)(A + (size_t)(mtile * 16 + (lane & 15)) * H);
  int kb0 = lane >> 4;   // lane<16 -> K {0..7,16..23}; lane>=16 -> {8..15,24..31}
  v16h a[4];
#pragma unroll
  for (int ks = 0; ks < 4; ++ks) {
    int cb = ks * 4 + kb0;
    v8h alo = arow[cb];
    v8h ahi = arow[cb + 2];
#pragma unroll
    for (int j = 0; j < 8; ++j) { a[ks][j] = alo[j]; a[ks][j + 8] = ahi[j]; }
  }

  float* Tw = T[wave];
#pragma unroll
  for (int half = 0; half < 2; ++half) {
    v8f acc[8];
#pragma unroll
    for (int i = 0; i < 8; ++i) acc[i] = v8f{0, 0, 0, 0, 0, 0, 0, 0};
#pragma unroll
    for (int ks = 0; ks < 4; ++ks) {
#pragma unroll
      for (int nt = 0; nt < 8; ++nt) {
        v16h b = ((const v16h*)Bw)[((half * 8 + nt) * 4 + ks) * 32 + lane];
        acc[nt] = __builtin_amdgcn_wmma_f32_16x16x32_f16(false, a[ks], false, b, (short)0,
                                                         acc[nt], false, false);
      }
    }
    // spill this half's C tiles to LDS (row M = r + 8*(lane>>4), col = nt*16+(lane&15))
#pragma unroll
    for (int nt = 0; nt < 8; ++nt)
#pragma unroll
      for (int r = 0; r < 8; ++r)
        Tw[(r + 8 * (lane >> 4)) * 264 + (half * 8 + nt) * 16 + (lane & 15)] = acc[nt][r];
  }
  __syncthreads();

  // LayerNorm over 256 cols: 2 lanes per row, 128 cols each
  int row16 = lane >> 1, colbase = (lane & 1) * 128;
  const float* R = Tw + row16 * 264 + colbase;
  float s = 0.0f, q = 0.0f;
  for (int c = 0; c < 128; c += 4) {
    float4 v = *(const float4*)(R + c);
    float4 bv = *(const float4*)(b1 + colbase + c);
    v.x += bv.x; v.y += bv.y; v.z += bv.z; v.w += bv.w;
    s += v.x + v.y + v.z + v.w;
    q += v.x * v.x + v.y * v.y + v.z * v.z + v.w * v.w;
  }
  s += __shfl_xor(s, 1);
  q += __shfl_xor(q, 1);
  float mu = s * (1.0f / H2);
  float var = q * (1.0f / H2) - mu * mu;
  float rs = rsqrtf(var + 1e-5f);

  _Float16* urow = U + (size_t)(mtile * 16 + row16) * H2 + colbase;
  for (int c = 0; c < 128; c += 4) {
    float4 v = *(const float4*)(R + c);
    float4 bv = *(const float4*)(b1 + colbase + c);
    float4 gg = *(const float4*)(g1 + colbase + c);
    float4 b2v = *(const float4*)(bb1 + colbase + c);
    urow[c + 0] = (_Float16)fmaxf((v.x + bv.x - mu) * rs * gg.x + b2v.x, 0.0f);
    urow[c + 1] = (_Float16)fmaxf((v.y + bv.y - mu) * rs * gg.y + b2v.y, 0.0f);
    urow[c + 2] = (_Float16)fmaxf((v.z + bv.z - mu) * rs * gg.z + b2v.z, 0.0f);
    urow[c + 3] = (_Float16)fmaxf((v.w + bv.w - mu) * rs * gg.w + b2v.w, 0.0f);
  }
}

// GEMM2: h = [h +] u @ W2 + b2 ; one wave per 16x16 output tile, K=256
__global__ void k_gemm2(const _Float16* __restrict__ U, const _Float16* __restrict__ Bw,
                        const float* __restrict__ b2, float* __restrict__ h, int residual) {
  int wave = threadIdx.x >> 5, lane = threadIdx.x & 31;
  int tile = blockIdx.x * 4 + wave;
  int mtile = tile >> 3, ntile = tile & 7;
  v8f acc = v8f{0, 0, 0, 0, 0, 0, 0, 0};
  const v8h* arow = (const v8h*)(U + (size_t)(mtile * 16 + (lane & 15)) * H2);
  int kb0 = lane >> 4;
#pragma unroll
  for (int ks = 0; ks < 8; ++ks) {
    int cb = ks * 4 + kb0;
    v8h alo = arow[cb];
    v8h ahi = arow[cb + 2];
    v16h a;
#pragma unroll
    for (int j = 0; j < 8; ++j) { a[j] = alo[j]; a[j + 8] = ahi[j]; }
    v16h b = ((const v16h*)Bw)[(ntile * 8 + ks) * 32 + lane];
    acc = __builtin_amdgcn_wmma_f32_16x16x32_f16(false, a, false, b, (short)0, acc, false, false);
  }
  int col = ntile * 16 + (lane & 15);
  float bias = b2[col];
#pragma unroll
  for (int r = 0; r < 8; ++r) {
    int grow = mtile * 16 + r + 8 * (lane >> 4);
    float v = acc[r] + bias;
    if (residual) v += h[(size_t)grow * H + col];
    h[(size_t)grow * H + col] = v;
  }
}

// final: out = hn @ lin_w + lin_b  (N x 16)
__global__ void k_out_linear(const float* __restrict__ hn, const float* __restrict__ lw,
                             const float* __restrict__ lb, float* __restrict__ out) {
  int tid = blockIdx.x * 256 + threadIdx.x;
  if (tid >= NN * DOUT) return;
  int o = tid & 15, n = tid >> 4;
  float acc = lb[o];
  const float* r = hn + (size_t)n * H;
  for (int c = 0; c < H; ++c) acc += r[c] * lw[c * DOUT + o];
  out[tid] = acc;
}

extern "C" void kernel_launch(void* const* d_in, const int* in_sizes, int n_in,
                              void* d_out, int out_size, void* d_ws, size_t ws_size,
                              hipStream_t stream) {
  const float* x     = (const float*)d_in[0];
  const int*   ei    = (const int*)d_in[1];
  const float* eattr = (const float*)d_in[2];
  const float* enw   = (const float*)d_in[3];
  const float* enb   = (const float*)d_in[4];
  const float* eew   = (const float*)d_in[5];
  const float* eeb   = (const float*)d_in[6];
  const float* t     = (const float*)d_in[7];
  const float* w1    = (const float*)d_in[8];
  const float* b1    = (const float*)d_in[9];
  const float* g1    = (const float*)d_in[10];
  const float* bb1   = (const float*)d_in[11];
  const float* w2    = (const float*)d_in[12];
  const float* b2    = (const float*)d_in[13];
  const float* lng   = (const float*)d_in[14];
  const float* lnb   = (const float*)d_in[15];
  const float* lw    = (const float*)d_in[16];
  const float* lb    = (const float*)d_in[17];

  char* ws = (char*)d_ws;
  const size_t NHb = (size_t)NN * H * 4;            // 10,240,000 B
  unsigned* mx      = (unsigned*)(ws + 0 * NHb);
  float*    den     = (float*)(ws + 1 * NHb);
  float*    num     = (float*)(ws + 2 * NHb);
  float*    hn      = (float*)(ws + 3 * NHb);
  float*    h       = (float*)(ws + 4 * NHb);
  _Float16* outf16  = (_Float16*)(ws + 5 * NHb);
  _Float16* u       = (_Float16*)(ws + 5 * NHb + (size_t)NN * H * 2);
  _Float16* w1h     = (_Float16*)(ws + 5 * NHb + (size_t)NN * H * 2 + (size_t)NN * H2 * 2);
  _Float16* w2h     = w1h + (size_t)NL * 32768;

  k_pack_weights<<<3072, 256, 0, stream>>>(w1, w2, w1h, w2h);
  k_zero<<<30000, 256, 0, stream>>>(mx, 3 * NN * H);   // mx,den,num contiguous
  k_encode_nodes<<<(NN * H) / 256, 256, 0, stream>>>(x, enw, enb, h);

  const int eg = (NE * H) / 256;  // 320000 blocks
  for (int i = 0; i < NL; ++i) {
    const float* hsrc;
    if (i == 0) {
      hsrc = h;   // first layer: conv applied directly to encoded features
    } else {
      k_prenorm<<<2500, 256, 0, stream>>>(h, lng + i * H, lnb + i * H, hn);
      hsrc = hn;
    }
    k_edge_max<<<eg, 256, 0, stream>>>(ei, eattr, hsrc, eew, eeb, t + i, mx);
    k_edge_sum<<<eg, 256, 0, stream>>>(ei, eattr, hsrc, eew, eeb, t + i, mx, den, num);
    k_node_finish<<<(NN * H) / 256, 256, 0, stream>>>(hsrc, num, den, mx, outf16);
    k_gemm1_ln_relu<<<625, 64, 0, stream>>>(outf16, w1h + (size_t)i * 32768,
                                            b1 + i * H2, g1 + i * H2, bb1 + i * H2, u);
    k_gemm2<<<2500, 128, 0, stream>>>(u, w2h + (size_t)i * 32768, b2 + i * H, h,
                                      (i > 0) ? 1 : 0);
  }

  k_prenorm<<<2500, 256, 0, stream>>>(h, lng, lnb, hn);   // layers[0] norm/act
  k_out_linear<<<(NN * DOUT + 255) / 256, 256, 0, stream>>>(hn, lw, lb, (float*)d_out);
}
